// MultiQueryAttention_48576080118370
// MI455X (gfx1250) — matrix operand order repaired and checked by reading, not compiled, not run on hardware
//
#include <hip/hip_runtime.h>

// ---------------------------------------------------------------------------
// MQA forward for MI455X (gfx1250, wave32, WMMA).
//   q = x@Wq+bq ; k = x@Wk+bk ; v = x@Wv+bv
//   attn = softmax(causal(q k^T / sqrt(Dh))) ; out = (attn v) @ Wo + bo
// All matmuls run through v_wmma_f32_16x16x32_bf16 (f32 accumulation).
// Attention is flash-style (online softmax) -> no B*H*T*T score tensor.
// GEMM: 64x64 tile per wave (16 WMMA : 16 b128 loads per K-step) + prefetch.
// Flash: 32 queries per wave; K/V tiles streamed into LDS by the Tensor Data
// Mover (tensor_load_to_lds, double-buffered, s_wait_tensorcnt) when the
// builtin is available, else direct global loads.
// ---------------------------------------------------------------------------

#define D_MODEL  2048
#define N_HEADS  16
#define HEAD_DIM 128
#define BATCH    4
#define SEQ      2048
#define M_TOTAL  (BATCH * SEQ)   // 8192 rows of (b,t)

typedef __bf16 bf16;
typedef __attribute__((ext_vector_type(16))) __bf16 v16bf;
typedef __attribute__((ext_vector_type(8)))  __bf16 v8bf;
typedef __attribute__((ext_vector_type(8)))  float  v8f;
typedef __attribute__((ext_vector_type(4)))  unsigned int u32x4;
typedef __attribute__((ext_vector_type(4)))  int i32x4;
typedef __attribute__((ext_vector_type(8)))  int i32x8;

#if __has_builtin(__builtin_amdgcn_tensor_load_to_lds) && \
    __has_builtin(__builtin_amdgcn_s_wait_tensorcnt)
#define USE_TDM 1
#endif

static __device__ __forceinline__ v8f v8f_zero() {
  v8f z;
#pragma unroll
  for (int i = 0; i < 8; ++i) z[i] = 0.0f;
  return z;
}

static __device__ __forceinline__ v8f wmma_bf16(v16bf a, v16bf b, v8f c) {
  // D = A(16x32 bf16) * B(32x16 bf16) + C(16x16 f32)
  return __builtin_amdgcn_wmma_f32_16x16x32_bf16(
      /*neg_a=*/false, a, /*neg_b=*/false, b,
      /*c_mod=*/(short)0, c, /*reuse_a=*/false, /*reuse_b=*/false);
}

// A-matrix 16x32 bf16 fragment (contiguous k order, same as B): lane holds row M = lane&15.
// elements 0..7  : K = half*16 + i     (half = lane>>4)
// elements 8..15 : K = half*16 + 8 + i
static __device__ __forceinline__ v16bf load_a_frag(const bf16* rowptr, int half) {
  v8bf lo = *(const v8bf*)(rowptr + half * 16);
  v8bf hi = *(const v8bf*)(rowptr + half * 16 + 8);
  v16bf r;
#pragma unroll
  for (int i = 0; i < 8; ++i) { r[i] = lo[i]; r[i + 8] = hi[i]; }
  return r;
}

#ifdef USE_TDM
// Issue a TDM 2D tile load (bf16 elements) into LDS. Descriptor per ISA
// cdna5 §8.3/8.4: group0 = {flags, lds_addr, global_addr, type}, group1 =
// {mask/data_size, dims, tile dims, dim0 stride}. All values wave-uniform.
static __device__ __forceinline__ void tdm_load_2d(
    const void* gptr,            // global address of tile start
    unsigned lds_byte_off,       // LDS byte offset of tile start
    unsigned row_elems,          // tensor_dim0 (elements per tensor row)
    unsigned rows_total,         // tensor_dim1 (rows in tensor)
    unsigned tile_d0,            // tile width (elements)
    unsigned tile_d1,            // tile height (rows)
    unsigned long long row_stride_elems) {  // tensor_dim0_stride
  unsigned long long ga = (unsigned long long)gptr;
  u32x4 g0;
  g0[0] = 1u;                                              // count=1, user D#
  g0[1] = lds_byte_off;                                    // lds_addr
  g0[2] = (unsigned)ga;                                    // global_addr[31:0]
  g0[3] = (unsigned)((ga >> 32) & 0x01ffffffu) | (2u << 30);  // addr[56:32]|type=2
  i32x8 g1;
  g1[0] = (int)(1u << 16);                                 // mask=0, data_size=2B
  g1[1] = (int)((row_elems & 0xffffu) << 16);              // tensor_dim0[15:0]
  g1[2] = (int)((row_elems >> 16) | ((rows_total & 0xffffu) << 16));
  g1[3] = (int)((rows_total >> 16) | (tile_d0 << 16));     // dim1 hi | tile_dim0
  g1[4] = (int)(tile_d1 & 0xffffu);                        // tile_dim1, tile_dim2=0
  g1[5] = (int)(unsigned)(row_stride_elems & 0xffffffffu); // dim0_stride lo
  g1[6] = (int)(unsigned)((row_stride_elems >> 32) & 0xffffu);  // stride hi
  g1[7] = 0;
  i32x4 gz = {0, 0, 0, 0};                                 // groups 2/3: 2D tensor
#if __clang_major__ >= 23
  i32x8 gz8 = {0, 0, 0, 0, 0, 0, 0, 0};
  __builtin_amdgcn_tensor_load_to_lds(g0, g1, gz, gz, gz8, 0);
#else
  __builtin_amdgcn_tensor_load_to_lds(g0, g1, gz, gz, 0);
#endif
}
#endif  // USE_TDM

// ---------------------------------------------------------------------------
// f32 -> bf16 elementwise convert (grid-stride)
// ---------------------------------------------------------------------------
__global__ void f32_to_bf16_kernel(const float* __restrict__ src,
                                   bf16* __restrict__ dst, size_t n) {
  size_t i = (size_t)blockIdx.x * blockDim.x + threadIdx.x;
  size_t stride = (size_t)gridDim.x * blockDim.x;
  for (; i < n; i += stride) dst[i] = (bf16)src[i];
}

// ---------------------------------------------------------------------------
// f32 KxN -> bf16 NxK transpose-convert (weights become N-major so WMMA
// B-fragments are contiguous 32B loads). 32x32 tiles via LDS.
// ---------------------------------------------------------------------------
__global__ __launch_bounds__(256)
void transpose_f32_bf16(const float* __restrict__ src, bf16* __restrict__ dst,
                        int K, int N) {
  __shared__ float tile[32][33];
  int n0 = blockIdx.x * 32, k0 = blockIdx.y * 32;
  int tx = threadIdx.x, ty = threadIdx.y;  // 32 x 8
#pragma unroll
  for (int j = 0; j < 32; j += 8)
    tile[ty + j][tx] = src[(size_t)(k0 + ty + j) * N + n0 + tx];
  __syncthreads();
#pragma unroll
  for (int j = 0; j < 32; j += 8)
    dst[(size_t)(n0 + ty + j) * K + k0 + tx] = (bf16)tile[tx][ty + j];
}

// ---------------------------------------------------------------------------
// bf16 WMMA GEMM:  C[M,N] = A[M,K] * Bt[N,K]^T + bias[N]
//   A  : bf16 row-major          Bt : bf16 N-major (i.e. B transposed)
//   mode 0: C = f32 row-major
//   mode 1: C = bf16 row-major
//   mode 2: C = bf16 stored [b][n][t] with m = b*Tdim + t   (V-transpose)
// Block = 128 threads (4 waves, 2x2). Block tile 128x128; wave tile 64x64
// (16 accumulators). Per K-step (32): 8 A-loads + 8 B-loads feed 16 WMMAs.
// ---------------------------------------------------------------------------
__global__ __launch_bounds__(128)
void gemm_bf16_wmma(const bf16* __restrict__ A, const bf16* __restrict__ Bt,
                    const float* __restrict__ bias, void* __restrict__ Cout,
                    int M, int N, int K, int mode, int Tdim) {
  const int lane = threadIdx.x & 31;
  const int wave = threadIdx.x >> 5;
  const int lm = lane & 15;
  const int half = lane >> 4;
  const int wm = wave >> 1, wn = wave & 1;
  const int m0 = blockIdx.y * 128 + wm * 64;
  const int n0 = blockIdx.x * 128 + wn * 64;

  v8f acc[4][4];
#pragma unroll
  for (int mt = 0; mt < 4; ++mt)
#pragma unroll
    for (int nt = 0; nt < 4; ++nt) acc[mt][nt] = v8f_zero();

  const bf16* arow0 = A + (size_t)(m0 + lm) * K;
  const bf16* brow0 = Bt + (size_t)(n0 + lm) * K + half * 16;

  for (int k0 = 0; k0 < K; k0 += 32) {
    // gfx1250 near-scope prefetch (global_prefetch_b8), ~1KB ahead
    __builtin_prefetch(arow0 + k0 + 512, 0, 3);
    __builtin_prefetch(brow0 + k0 + 512, 0, 3);

    v16bf af[4];
#pragma unroll
    for (int mt = 0; mt < 4; ++mt)
      af[mt] = load_a_frag(arow0 + (size_t)mt * 16 * K + k0, half);
#pragma unroll
    for (int nt = 0; nt < 4; ++nt) {
      // B fragment: lane -> col n = lm; elements i -> k = half*16 + i
      v16bf bfrag = *(const v16bf*)(brow0 + (size_t)nt * 16 * K + k0);
#pragma unroll
      for (int mt = 0; mt < 4; ++mt)
        acc[mt][nt] = wmma_bf16(af[mt], bfrag, acc[mt][nt]);
    }
  }

#pragma unroll
  for (int nt = 0; nt < 4; ++nt) {
    const int n = n0 + nt * 16 + lm;
    const float bv = bias[n];
#pragma unroll
    for (int mt = 0; mt < 4; ++mt) {
#pragma unroll
      for (int r = 0; r < 8; ++r) {
        const int m = m0 + mt * 16 + r + 8 * half;  // C/D layout: row = r + 8*half
        const float v = acc[mt][nt][r] + bv;
        if (mode == 0) {
          ((float*)Cout)[(size_t)m * N + n] = v;
        } else if (mode == 1) {
          ((bf16*)Cout)[(size_t)m * N + n] = (bf16)v;
        } else {
          const int b = m / Tdim, t = m % Tdim;
          ((bf16*)Cout)[((size_t)b * N + n) * Tdim + t] = (bf16)v;
        }
      }
    }
  }
}

// ---------------------------------------------------------------------------
// Flash attention, multi-query (one shared K/V head).
// One wave per (b, h, 32-query tile): two 16-row q-tiles with independent
// online-softmax state; every K- and V-fragment is used by both q-tiles.
// K/V tiles for each 32-key block are streamed into LDS by the TDM
// (tensor_load_to_lds), double-buffered so block s+1 loads while block s
// computes; completion is gated with s_wait_tensorcnt.
// ---------------------------------------------------------------------------
__global__ __launch_bounds__(32)
void flash_mqa(const bf16* __restrict__ Q, const bf16* __restrict__ Kc,
               const bf16* __restrict__ Vt, bf16* __restrict__ Oatt) {
  const int nqt = SEQ / 32;
  const int qt = blockIdx.x % nqt;
  const int h  = (blockIdx.x / nqt) % N_HEADS;
  const int b  = blockIdx.x / (nqt * N_HEADS);
  const int lane = threadIdx.x;
  const int lm = lane & 15, half = lane >> 4;
  const int q0 = qt * 32;                    // q-tile t covers rows q0+t*16 ..+15
  const float scale = 0.08838834764831845f;  // 1/sqrt(128)

  __shared__ __attribute__((aligned(32))) bf16 pbuf[32 * 32];           // 2 KB
#ifdef USE_TDM
  __shared__ __attribute__((aligned(64))) bf16 ldsK[2][32 * HEAD_DIM];  // 2x8 KB
  __shared__ __attribute__((aligned(64))) bf16 ldsV[2][HEAD_DIM * 32];  // 2x8 KB
#endif

  // Q fragments: [q-tile][f], f covers Dh contraction chunk [f*32, f*32+32)
  v16bf qf[2][4];
#pragma unroll
  for (int t = 0; t < 2; ++t) {
    const bf16* qrow =
        Q + (size_t)(b * SEQ + q0 + t * 16 + lm) * D_MODEL + h * HEAD_DIM;
#pragma unroll
    for (int f = 0; f < 4; ++f) qf[t][f] = load_a_frag(qrow + f * 32, half);
  }

  v8f o[2][8];
  float mi[2][8], li[2][8];
#pragma unroll
  for (int t = 0; t < 2; ++t)
#pragma unroll
    for (int r = 0; r < 8; ++r) {
      o[t][r] = v8f_zero();
      mi[t][r] = -3.0e38f;
      li[t][r] = 0.0f;
    }

#ifdef USE_TDM
  // TDM tile issue: K tile = 32 keys x 128 dh (row-major in LDS),
  //                 V tile = 128 d-rows x 32 keys (Vt is [b][d][t]).
  auto issue_tiles = [&](int s0, int buf) {
    tdm_load_2d(Kc + (size_t)(b * SEQ + s0) * HEAD_DIM,
                (unsigned)(unsigned long long)&ldsK[buf][0],
                HEAD_DIM, (unsigned)M_TOTAL, HEAD_DIM, 32,
                (unsigned long long)HEAD_DIM);
    tdm_load_2d(Vt + (size_t)b * HEAD_DIM * SEQ + s0,
                (unsigned)(unsigned long long)&ldsV[buf][0],
                SEQ, (unsigned)(BATCH * HEAD_DIM), 32, HEAD_DIM,
                (unsigned long long)SEQ);
  };
  issue_tiles(0, 0);
#endif
  int cur = 0;

  for (int s0 = 0; s0 <= q0 + 31; s0 += 32) {  // causal: skip fully-masked blocks
#ifdef USE_TDM
    if (s0 + 32 <= q0 + 31) {
      issue_tiles(s0 + 32, cur ^ 1);           // overlap next block's DMA
      __builtin_amdgcn_s_wait_tensorcnt(2);    // current buffer complete
    } else {
      __builtin_amdgcn_s_wait_tensorcnt(0);
    }
#endif
    __syncthreads();  // order LDS traffic (pbuf reuse + TDM-written tiles)

    // ---- S = Q K^T for keys [s0, s0+32): 2 q-tiles x 2 key-tiles ----
    v8f s[2][2];
#pragma unroll
    for (int t = 0; t < 2; ++t) { s[t][0] = v8f_zero(); s[t][1] = v8f_zero(); }
#pragma unroll
    for (int f = 0; f < 4; ++f) {
#ifdef USE_TDM
      const bf16* kbase = &ldsK[cur][f * 32 + half * 16];
      v16bf kb0 = *(const v16bf*)(kbase + lm * HEAD_DIM);
      v16bf kb1 = *(const v16bf*)(kbase + (16 + lm) * HEAD_DIM);
#else
      const bf16* kp0 =
          Kc + (size_t)(b * SEQ + s0 + lm) * HEAD_DIM + f * 32 + half * 16;
      v16bf kb0 = *(const v16bf*)kp0;
      v16bf kb1 = *(const v16bf*)(kp0 + (size_t)16 * HEAD_DIM);
#endif
      s[0][0] = wmma_bf16(qf[0][f], kb0, s[0][0]);
      s[0][1] = wmma_bf16(qf[0][f], kb1, s[0][1]);
      s[1][0] = wmma_bf16(qf[1][f], kb0, s[1][0]);
      s[1][1] = wmma_bf16(qf[1][f], kb1, s[1][1]);
    }

    // ---- scale + causal mask + online softmax (C-layout), both q-tiles ----
    const int k0i = s0 + lm, k1i = s0 + 16 + lm;
#pragma unroll
    for (int t = 0; t < 2; ++t) {
      float p0[8], p1[8], alpha[8];
#pragma unroll
      for (int r = 0; r < 8; ++r) {
        const int qq = q0 + t * 16 + r + 8 * half;
        p0[r] = (k0i <= qq) ? s[t][0][r] * scale : -3.0e38f;
        p1[r] = (k1i <= qq) ? s[t][1][r] * scale : -3.0e38f;
        float mx = fmaxf(p0[r], p1[r]);
#pragma unroll
        for (int off = 1; off < 16; off <<= 1)  // reduce over 16 key lanes
          mx = fmaxf(mx, __shfl_xor(mx, off, 16));
        const float mnew = fmaxf(mi[t][r], mx);
        alpha[r] = __expf(mi[t][r] - mnew);
        mi[t][r] = mnew;
        p0[r] = __expf(p0[r] - mnew);
        p1[r] = __expf(p1[r] - mnew);
        float ls = p0[r] + p1[r];
#pragma unroll
        for (int off = 1; off < 16; off <<= 1)
          ls += __shfl_xor(ls, off, 16);
        li[t][r] = li[t][r] * alpha[r] + ls;
      }
#pragma unroll
      for (int dt = 0; dt < 8; ++dt)
#pragma unroll
        for (int r = 0; r < 8; ++r) o[t][dt][r] *= alpha[r];
      // P tile t -> LDS rows [t*16, t*16+16), row-major 16x32 bf16
#pragma unroll
      for (int r = 0; r < 8; ++r) {
        pbuf[(t * 16 + r + 8 * half) * 32 + lm]      = (bf16)p0[r];
        pbuf[(t * 16 + r + 8 * half) * 32 + 16 + lm] = (bf16)p1[r];
      }
    }
    __syncthreads();

    // ---- reload P in A-layout (cross-lane transpose via LDS) ----
    v16bf pf[2];
#pragma unroll
    for (int t = 0; t < 2; ++t) {
      v8bf lo = *(const v8bf*)&pbuf[(t * 16 + lm) * 32 + half * 16];
      v8bf hi = *(const v8bf*)&pbuf[(t * 16 + lm) * 32 + half * 16 + 8];
#pragma unroll
      for (int i = 0; i < 8; ++i) { pf[t][i] = lo[i]; pf[t][i + 8] = hi[i]; }
    }

    // ---- O += P V : each V fragment feeds both q-tiles ----
#pragma unroll
    for (int dt = 0; dt < 8; ++dt) {
#ifdef USE_TDM
      v16bf vb = *(const v16bf*)&ldsV[cur][(dt * 16 + lm) * 32 + half * 16];
#else
      const bf16* vp =
          Vt + ((size_t)b * HEAD_DIM + dt * 16 + lm) * SEQ + s0 + half * 16;
      v16bf vb = *(const v16bf*)vp;
#endif
      o[0][dt] = wmma_bf16(pf[0], vb, o[0][dt]);
      o[1][dt] = wmma_bf16(pf[1], vb, o[1][dt]);
    }
    cur ^= 1;
  }

  // ---- epilogue: O /= l, store bf16 row-major for the Wo projection ----
#pragma unroll
  for (int t = 0; t < 2; ++t) {
    float invl[8];
#pragma unroll
    for (int r = 0; r < 8; ++r) invl[r] = 1.0f / li[t][r];
#pragma unroll
    for (int dt = 0; dt < 8; ++dt) {
#pragma unroll
      for (int r = 0; r < 8; ++r) {
        const int qq = q0 + t * 16 + r + 8 * half;
        Oatt[(size_t)(b * SEQ + qq) * D_MODEL + h * HEAD_DIM + dt * 16 + lm] =
            (bf16)(o[t][dt][r] * invl[r]);
      }
    }
  }
}

// ---------------------------------------------------------------------------
// Host launcher. Workspace layout (~123 MB of d_ws):
//   xb, WqT, WkT, WvT, WoT, Qb, Kb, Vt, attn_b  (all bf16, 256B aligned)
// ---------------------------------------------------------------------------
extern "C" void kernel_launch(void* const* d_in, const int* in_sizes, int n_in,
                              void* d_out, int out_size, void* d_ws, size_t ws_size,
                              hipStream_t stream) {
  (void)in_sizes; (void)n_in; (void)out_size; (void)ws_size;
  const float* x  = (const float*)d_in[0];
  // d_in[1] is the causal mask; applied analytically inside flash_mqa.
  const float* Wq = (const float*)d_in[2];
  const float* bq = (const float*)d_in[3];
  const float* Wk = (const float*)d_in[4];
  const float* bk = (const float*)d_in[5];
  const float* Wv = (const float*)d_in[6];
  const float* bv = (const float*)d_in[7];
  const float* Wo = (const float*)d_in[8];
  const float* bo = (const float*)d_in[9];
  float* out = (float*)d_out;

  char* ws = (char*)d_ws;
  size_t off = 0;
  auto wsAlloc = [&](size_t elems) -> bf16* {
    bf16* p = (bf16*)(ws + off);
    off += ((elems * sizeof(bf16) + 255) & ~(size_t)255);
    return p;
  };
  bf16* xb  = wsAlloc((size_t)M_TOTAL * D_MODEL);
  bf16* WqT = wsAlloc((size_t)D_MODEL * D_MODEL);
  bf16* WkT = wsAlloc((size_t)HEAD_DIM * D_MODEL);
  bf16* WvT = wsAlloc((size_t)HEAD_DIM * D_MODEL);
  bf16* WoT = wsAlloc((size_t)D_MODEL * D_MODEL);
  bf16* Qb  = wsAlloc((size_t)M_TOTAL * D_MODEL);
  bf16* Kb  = wsAlloc((size_t)M_TOTAL * HEAD_DIM);
  bf16* Vt  = wsAlloc((size_t)BATCH * HEAD_DIM * SEQ);
  bf16* Ab  = wsAlloc((size_t)M_TOTAL * D_MODEL);

  // 1) casts / weight transposes
  f32_to_bf16_kernel<<<4096, 256, 0, stream>>>(x, xb, (size_t)M_TOTAL * D_MODEL);
  dim3 tb(32, 8);
  transpose_f32_bf16<<<dim3(D_MODEL / 32, D_MODEL / 32), tb, 0, stream>>>(Wq, WqT, D_MODEL, D_MODEL);
  transpose_f32_bf16<<<dim3(HEAD_DIM / 32, D_MODEL / 32), tb, 0, stream>>>(Wk, WkT, D_MODEL, HEAD_DIM);
  transpose_f32_bf16<<<dim3(HEAD_DIM / 32, D_MODEL / 32), tb, 0, stream>>>(Wv, WvT, D_MODEL, HEAD_DIM);
  transpose_f32_bf16<<<dim3(D_MODEL / 32, D_MODEL / 32), tb, 0, stream>>>(Wo, WoT, D_MODEL, D_MODEL);

  // 2) projections (WMMA)
  gemm_bf16_wmma<<<dim3(D_MODEL / 128, M_TOTAL / 128), 128, 0, stream>>>(
      xb, WqT, bq, Qb, M_TOTAL, D_MODEL, D_MODEL, /*mode=*/1, SEQ);
  gemm_bf16_wmma<<<dim3(HEAD_DIM / 128, M_TOTAL / 128), 128, 0, stream>>>(
      xb, WkT, bk, Kb, M_TOTAL, HEAD_DIM, D_MODEL, /*mode=*/1, SEQ);
  gemm_bf16_wmma<<<dim3(HEAD_DIM / 128, M_TOTAL / 128), 128, 0, stream>>>(
      xb, WvT, bv, Vt, M_TOTAL, HEAD_DIM, D_MODEL, /*mode=*/2, SEQ);

  // 3) flash attention (WMMA + TDM double-buffered K/V tiles)
  flash_mqa<<<BATCH * N_HEADS * (SEQ / 32), 32, 0, stream>>>(Qb, Kb, Vt, Ab);

  // 4) output projection (WMMA, f32 out)
  gemm_bf16_wmma<<<dim3(D_MODEL / 128, M_TOTAL / 128), 128, 0, stream>>>(
      Ab, WoT, bo, out, M_TOTAL, D_MODEL, D_MODEL, /*mode=*/0, SEQ);
}
